// AttentionMap_59957743452239
// MI455X (gfx1250) — matrix operand order repaired and checked by reading, not compiled, not run on hardware
//
#include <hip/hip_runtime.h>

// ---------------------------------------------------------------------------
// Attention: out = softmax( (Xg Wq + bq)(Xl Wk + bk)^T / 16 ) (Xl Wv + bv)
// B=4, SQ=SKV=4096, DIN=DK=256, f32 in / f32 out.
// bf16 WMMA (f32 acc) flash attention + Tensor Data Mover double-buffered
// K/V tile staging into LDS (TENSORcnt-tracked async DMA).
// ---------------------------------------------------------------------------

typedef __attribute__((ext_vector_type(16))) __bf16 v16bf;
typedef __attribute__((ext_vector_type(8)))  __bf16 v8bf;
typedef __attribute__((ext_vector_type(8)))  float  v8f;
typedef __attribute__((ext_vector_type(4)))  unsigned int v4u;
typedef __attribute__((ext_vector_type(8)))  int v8i;
typedef __attribute__((ext_vector_type(4)))  int v4i;

#define BATCH 4
#define SEQ   4096
#define DIM   256
#define SCALE 0.0625f   // 1/sqrt(256)

static __device__ __forceinline__ v16bf cat8(v8bf lo, v8bf hi) {
    v16bf r;
#pragma unroll
    for (int i = 0; i < 8; ++i) { r[i] = lo[i]; r[i + 8] = hi[i]; }
    return r;
}

// TDM descriptor group 0: count=1, lds byte addr, global byte addr, type=2.
static __device__ __forceinline__ v4u tdm_g0(unsigned lds_addr, const void* gptr) {
    unsigned long long ga = (unsigned long long)(uintptr_t)gptr;
    v4u g0;
    g0[0] = 1u;                                   // count=1 (valid descriptor)
    g0[1] = lds_addr;                             // bits 63:32 lds_addr
    g0[2] = (unsigned)ga;                         // global_addr[31:0]
    g0[3] = (unsigned)((ga >> 32) & 0x1FFFFFFu)   // global_addr[56:32]
          | 0x80000000u;                          // type=2 ("image")
    return g0;
}

// ---------------------------------------------------------------------------
// W (f32 [DIN][DK]) -> Wt (bf16 [DK][DIN]) so proj B-fragments are contiguous.
// ---------------------------------------------------------------------------
__global__ __launch_bounds__(256)
void transpose_w_bf16(const float* __restrict__ W, __bf16* __restrict__ Wt)
{
    const int c = blockIdx.x;      // output row (= W column)
    const int r = threadIdx.x;     // output col (= W row)
    Wt[(size_t)c * DIM + r] = (__bf16)W[(size_t)r * DIM + c];
}

// ---------------------------------------------------------------------------
// Phase 1: per-batch GEMM  X[4096x256] * W[256x256] + b  -> bf16
// A from f32 global (cvt), B from bf16 Wt (contiguous). 8 WMMA per tile.
// ---------------------------------------------------------------------------
__global__ __launch_bounds__(256)
void proj_gemm_bf16(const float* __restrict__ X, const __bf16* __restrict__ Wt,
                    const float* __restrict__ bias, __bf16* __restrict__ out,
                    int transposeOut)
{
    const int bat  = blockIdx.z;
    const int r0   = blockIdx.x * 16;
    const int wave = threadIdx.x >> 5;
    const int lane = threadIdx.x & 31;
    const int ln   = lane & 15;
    const int hi   = lane >> 4;
    const int colBase = (blockIdx.y * 8 + wave) * 16;

    const float*  xr = X  + ((size_t)bat * SEQ + r0 + ln) * DIM;
    const __bf16* wt = Wt + (size_t)(colBase + ln) * DIM;

    v8f acc = {0.f, 0.f, 0.f, 0.f, 0.f, 0.f, 0.f, 0.f};

#pragma unroll
    for (int c = 0; c < 8; ++c) {
        const float* p0 = xr + c * 32 + 8 * hi;
        const float* p1 = xr + c * 32 + 16 + 8 * hi;
        v16bf af;
#pragma unroll
        for (int i = 0; i < 8; ++i) {
            af[i]     = (__bf16)p0[i];
            af[i + 8] = (__bf16)p1[i];
        }
        const __bf16* wp = wt + c * 32 + 8 * hi;
        v16bf bf = cat8(*(const v8bf*)wp, *(const v8bf*)(wp + 16));

        acc = __builtin_amdgcn_wmma_f32_16x16x32_bf16(false, af, false, bf,
                                                      (short)0, acc, false, false);
    }

    const float bcol = bias[colBase + ln];
    if (!transposeOut) {
#pragma unroll
        for (int v = 0; v < 8; ++v)
            out[((size_t)bat * SEQ + r0 + 8 * hi + v) * DIM + colBase + ln] =
                (__bf16)(acc[v] + bcol);
    } else {
#pragma unroll
        for (int v = 0; v < 8; ++v)
            out[((size_t)bat * DIM + colBase + ln) * SEQ + r0 + 8 * hi + v] =
                (__bf16)(acc[v] + bcol);
    }
}

// ---------------------------------------------------------------------------
// Phase 2: flash attention, TDM double-buffered K/V tiles.
// Block = 8 waves, 128 q-rows. Wave: 16 q-rows x 256 d, kv steps of 32.
// ---------------------------------------------------------------------------
__global__ __launch_bounds__(256)
void flash_attn_bf16(const __bf16* __restrict__ Qb, const __bf16* __restrict__ Kb,
                     const __bf16* __restrict__ Vt, float* __restrict__ Out)
{
    // [buf][kv][d] row padded 256->264 ; [buf][d][kv] row padded 32->40
    __shared__ __align__(16) __bf16 Ks[2][32 * 264];
    __shared__ __align__(16) __bf16 Vs[2][256 * 40];
    __shared__ __align__(16) __bf16 Ps[8 * 16 * 40];

    const int bat  = blockIdx.y;
    const int wave = threadIdx.x >> 5;
    const int lane = threadIdx.x & 31;
    const int ln   = lane & 15;
    const int hi   = lane >> 4;
    const int q0   = blockIdx.x * 128 + wave * 16;

    // D# group 1 constants (data_size=2B, pad_enable, 2-D tile):
    // K tile: 32 rows x 256 elems; pad 4 dwords every 128 dwords (row 512B->528B)
    const v8i Kg1 = { 0x07910000, (int)(256u << 16), (int)(4096u << 16),
                      (int)(256u << 16), 32, 256, 0, 0 };
    // V tile: 256 rows x 32 elems; pad 4 dwords every 16 dwords (row 64B->80B)
    const v8i Vg1 = { 0x06D10000, (int)(4096u << 16), (int)(256u << 16),
                      (int)(32u << 16), 256, 4096, 0, 0 };
    const v4i z4 = {0, 0, 0, 0};
    const v8i z8 = {0, 0, 0, 0, 0, 0, 0, 0};

    const __bf16* kbase = Kb + (size_t)bat * SEQ * DIM;
    const __bf16* vbase = Vt + (size_t)bat * DIM * SEQ;

    // Preload Q A-fragments (register resident).
    const __bf16* qrow = Qb + ((size_t)bat * SEQ + q0 + ln) * DIM;
    v16bf qf[8];
#pragma unroll
    for (int c = 0; c < 8; ++c)
        qf[c] = cat8(*(const v8bf*)(qrow + c * 32 + 8 * hi),
                     *(const v8bf*)(qrow + c * 32 + 16 + 8 * hi));

    v8f acc[16];
#pragma unroll
    for (int t = 0; t < 16; ++t)
#pragma unroll
        for (int v = 0; v < 8; ++v) acc[t][v] = 0.f;

    float rm[8], rs[8];
#pragma unroll
    for (int v = 0; v < 8; ++v) { rm[v] = -1e30f; rs[v] = 0.f; }

    __bf16* Pw = Ps + wave * (16 * 40);

    // Prologue: wave 0 kicks off the TDM loads for the first kv tile (buf 0).
    if (wave == 0) {
        __builtin_amdgcn_tensor_load_to_lds(
            tdm_g0((unsigned)(uintptr_t)&Ks[0][0], kbase), Kg1, z4, z4, z8, 0);
        __builtin_amdgcn_tensor_load_to_lds(
            tdm_g0((unsigned)(uintptr_t)&Vs[0][0], vbase), Vg1, z4, z4, z8, 0);
    }

    const int NIT = SEQ / 32;   // 128
    for (int i = 0; i < NIT; ++i) {
        const int cur = i & 1;
        const int kvn = (i + 1) * 32;

        __syncthreads();   // everyone done reading buf cur^1 (prev iteration)
        if (wave == 0) {
            if (i + 1 < NIT) {
                // prefetch next tile into the other buffer, then wait until the
                // two older TDM ops (current buffer) have completed (in-order).
                __builtin_amdgcn_tensor_load_to_lds(
                    tdm_g0((unsigned)(uintptr_t)&Ks[cur ^ 1][0],
                           kbase + (size_t)kvn * DIM), Kg1, z4, z4, z8, 0);
                __builtin_amdgcn_tensor_load_to_lds(
                    tdm_g0((unsigned)(uintptr_t)&Vs[cur ^ 1][0],
                           vbase + kvn), Vg1, z4, z4, z8, 0);
                __builtin_amdgcn_s_wait_tensorcnt(2);
            } else {
                __builtin_amdgcn_s_wait_tensorcnt(0);
            }
        }
        __syncthreads();   // current buffer visible to all waves

        // ---- S = Q K^T for two 16-kv column tiles ----
        v8f s0 = {0.f,0.f,0.f,0.f,0.f,0.f,0.f,0.f};
        v8f s1 = {0.f,0.f,0.f,0.f,0.f,0.f,0.f,0.f};
        const __bf16* k0 = &Ks[cur][0] + ln * 264;
        const __bf16* k1 = &Ks[cur][0] + (16 + ln) * 264;
#pragma unroll
        for (int c = 0; c < 8; ++c) {
            v16bf b0 = cat8(*(const v8bf*)(k0 + c * 32 + 8 * hi),
                            *(const v8bf*)(k0 + c * 32 + 16 + 8 * hi));
            v16bf b1 = cat8(*(const v8bf*)(k1 + c * 32 + 8 * hi),
                            *(const v8bf*)(k1 + c * 32 + 16 + 8 * hi));
            s0 = __builtin_amdgcn_wmma_f32_16x16x32_bf16(false, qf[c], false, b0,
                                                         (short)0, s0, false, false);
            s1 = __builtin_amdgcn_wmma_f32_16x16x32_bf16(false, qf[c], false, b1,
                                                         (short)0, s1, false, false);
        }

        // ---- online softmax (row = across a 16-lane half) ----
#pragma unroll
        for (int v = 0; v < 8; ++v) {
            float a0 = s0[v] * SCALE, a1 = s1[v] * SCALE;
            float mx = fmaxf(a0, a1);
            mx = fmaxf(mx, __shfl_xor(mx, 1, 32));
            mx = fmaxf(mx, __shfl_xor(mx, 2, 32));
            mx = fmaxf(mx, __shfl_xor(mx, 4, 32));
            mx = fmaxf(mx, __shfl_xor(mx, 8, 32));
            float mnew = fmaxf(rm[v], mx);
            float fct  = __expf(rm[v] - mnew);
            float p0   = __expf(a0 - mnew);
            float p1   = __expf(a1 - mnew);
            float psum = p0 + p1;
            psum += __shfl_xor(psum, 1, 32);
            psum += __shfl_xor(psum, 2, 32);
            psum += __shfl_xor(psum, 4, 32);
            psum += __shfl_xor(psum, 8, 32);
            rs[v] = rs[v] * fct + psum;
            rm[v] = mnew;
#pragma unroll
            for (int t = 0; t < 16; ++t) acc[t][v] *= fct;
            const int m = 8 * hi + v;
            Pw[m * 40 + ln]      = (__bf16)p0;
            Pw[m * 40 + 16 + ln] = (__bf16)p1;
        }

        // ---- O += P V (same-wave DS ops are in-order; no barrier needed) ----
        v16bf pa = cat8(*(const v8bf*)(Pw + ln * 40 + 8 * hi),
                        *(const v8bf*)(Pw + ln * 40 + 16 + 8 * hi));
#pragma unroll
        for (int t = 0; t < 16; ++t) {
            const __bf16* vr = &Vs[cur][0] + (t * 16 + ln) * 40 + 8 * hi;
            v16bf bv = cat8(*(const v8bf*)vr, *(const v8bf*)(vr + 16));
            acc[t] = __builtin_amdgcn_wmma_f32_16x16x32_bf16(false, pa, false, bv,
                                                             (short)0, acc[t], false, false);
        }
    }

    // ---- normalize and store f32 output ----
    float* orow = Out + ((size_t)bat * SEQ + q0) * DIM;
#pragma unroll
    for (int t = 0; t < 16; ++t) {
#pragma unroll
        for (int v = 0; v < 8; ++v)
            orow[(size_t)(8 * hi + v) * DIM + t * 16 + ln] = acc[t][v] / rs[v];
    }
}

// ---------------------------------------------------------------------------
extern "C" void kernel_launch(void* const* d_in, const int* in_sizes, int n_in,
                              void* d_out, int out_size, void* d_ws, size_t ws_size,
                              hipStream_t stream)
{
    const float* conv_local  = (const float*)d_in[0];
    const float* conv_global = (const float*)d_in[1];
    const float* Wk = (const float*)d_in[2];
    const float* bk = (const float*)d_in[3];
    const float* Wq = (const float*)d_in[4];
    const float* bq = (const float*)d_in[5];
    const float* Wv = (const float*)d_in[6];
    const float* bv = (const float*)d_in[7];

    // workspace: Q, K (bf16 row-major), V^T (bf16 [D][SKV]), 3x Wt (bf16)
    const size_t elems = (size_t)BATCH * SEQ * DIM;
    __bf16* Qb  = (__bf16*)d_ws;
    __bf16* Kb  = Qb + elems;
    __bf16* Vt  = Kb + elems;
    __bf16* Wtq = Vt + elems;
    __bf16* Wtk = Wtq + (size_t)DIM * DIM;
    __bf16* Wtv = Wtk + (size_t)DIM * DIM;

    transpose_w_bf16<<<DIM, 256, 0, stream>>>(Wq, Wtq);
    transpose_w_bf16<<<DIM, 256, 0, stream>>>(Wk, Wtk);
    transpose_w_bf16<<<DIM, 256, 0, stream>>>(Wv, Wtv);

    dim3 gproj(SEQ / 16, 2, BATCH);
    proj_gemm_bf16<<<gproj, 256, 0, stream>>>(conv_global, Wtq, bq, Qb, 0);
    proj_gemm_bf16<<<gproj, 256, 0, stream>>>(conv_local,  Wtk, bk, Kb, 0);
    proj_gemm_bf16<<<gproj, 256, 0, stream>>>(conv_local,  Wtv, bv, Vt, 1);

    dim3 gattn(SEQ / 128, BATCH);
    flash_attn_bf16<<<gattn, 256, 0, stream>>>(Qb, Kb, Vt, (float*)d_out);
}
